// GCN_PYG_83915071029568
// MI455X (gfx1250) — matrix operand order repaired and checked by reading, not compiled, not run on hardware
//
#include <hip/hip_runtime.h>

// ---------------- problem constants ----------------
constexpr int kBatch = 16;
constexpr int kN     = 1024;
constexpr int kFin   = 256;
constexpr int kHid   = 512;
constexpr int kOut   = 256;

// ---------------- WMMA tile config ----------------
#define BM 128
#define BN 128
#define BK 32
#define LDSPAD 8   // LDS row padding (elements)

typedef __attribute__((ext_vector_type(16))) __bf16 v16bf;
typedef __attribute__((ext_vector_type(8)))  __bf16 v8bf;
typedef __attribute__((ext_vector_type(8)))  float  v8f;
typedef __attribute__((ext_vector_type(4)))  int    v4i;

typedef __attribute__((address_space(1))) v4i* gptr_v4i;
typedef __attribute__((address_space(3))) v4i* lptr_v4i;

// ---------------- gfx1250 async global->LDS copy ----------------
#if defined(__has_builtin)
#if __has_builtin(__builtin_amdgcn_global_load_async_to_lds_b128)
#define GCN_ASYNC_LDS 1
#endif
#endif

__device__ inline void copy16_to_lds(const __bf16* gsrc, __bf16* ldst) {
#ifdef GCN_ASYNC_LDS
  __builtin_amdgcn_global_load_async_to_lds_b128(
      (gptr_v4i)(uintptr_t)gsrc,
      (lptr_v4i)(uint32_t)(uintptr_t)ldst, 0, 0);
#else
  *(uint4*)ldst = *(const uint4*)gsrc;
#endif
}

__device__ inline void wait_async_copies() {
#ifdef GCN_ASYNC_LDS
#if __has_builtin(__builtin_amdgcn_s_wait_asynccnt)
  __builtin_amdgcn_s_wait_asynccnt(0);
#else
  asm volatile("s_wait_asynccnt 0x0" ::: "memory");
#endif
#endif
}

// ---------------- prep kernels ----------------

__global__ void f32_to_bf16_kernel(const float* __restrict__ src,
                                   __bf16* __restrict__ dst, int n) {
  int i = blockIdx.x * blockDim.x + threadIdx.x;
  if (i < n) dst[i] = (__bf16)src[i];
}

// dinv[b,v] = rsqrt( sum_u A_hat[u,v] ), A_hat = max(adj>0, I)
__global__ void deg_kernel(const int* __restrict__ adj, float* __restrict__ dinv) {
  int idx = blockIdx.x * blockDim.x + threadIdx.x;  // b*kN + v
  int b = idx >> 10;
  int v = idx & (kN - 1);
  const int* col = adj + (long long)b * kN * kN + v;
  int cnt = 0;
#pragma unroll 4
  for (int u = 0; u < kN; ++u) {
    int a = col[(long long)u * kN];
    cnt += ((a > 0) || (u == v)) ? 1 : 0;
  }
  dinv[idx] = rsqrtf((float)cnt);
}

// ATs[b,v,u] = A_hat[u,v] * dinv[b,u]  (bf16, v-major) -- transpose via LDS
__global__ void build_ats_kernel(const int* __restrict__ adj,
                                 const float* __restrict__ dinv,
                                 __bf16* __restrict__ ats) {
  __shared__ __bf16 tile[32][34];
  const int b  = blockIdx.z;
  const int u0 = blockIdx.y * 32;
  const int v0 = blockIdx.x * 32;
  const int tx = threadIdx.x;   // 0..31
  const int ty = threadIdx.y;   // 0..7
  const long long base = (long long)b * kN * kN;
#pragma unroll
  for (int s = 0; s < 4; ++s) {
    int u = u0 + ty + s * 8;
    int a = adj[base + (long long)u * kN + v0 + tx];
    bool edge = (a > 0) || (u == (v0 + tx));
    tile[ty + s * 8][tx] = edge ? (__bf16)dinv[b * kN + u] : (__bf16)0.0f;
  }
  __syncthreads();
#pragma unroll
  for (int s = 0; s < 4; ++s) {
    int v = v0 + ty + s * 8;
    ats[base + (long long)v * kN + u0 + tx] = tile[tx][ty + s * 8];
  }
}

// out[b,f] = b2[f] + (1/N) * sum_n out2[b,n,f]
__global__ void pool_kernel(const float* __restrict__ out2,
                            const float* __restrict__ b2,
                            float* __restrict__ out) {
  int i = blockIdx.x * blockDim.x + threadIdx.x;   // kBatch*kOut
  int b = i >> 8;
  int f = i & (kOut - 1);
  const float* p = out2 + (long long)b * kN * kOut + f;
  float s = 0.0f;
#pragma unroll 4
  for (int n = 0; n < kN; ++n) s += p[(long long)n * kOut];
  out[i] = b2[f] + s * (1.0f / (float)kN);
}

// ---------------- main WMMA matmul ----------------
// C[b] = A[b] (M x K, bf16 row-major) @ B[b] (K x Nn, bf16 row-major)
// Block tile 128x128, 8 waves (4 M x 2 N), wave tile 32x64 (2x4 WMMA tiles).
// Double-buffered LDS; A tile copied via async global->LDS, B tile transposed.
// MODE 0: store bf16 acc
// MODE 1: store bf16 relu(dinv[m]*acc + bias[n])
// MODE 2: store f32  dinv[m]*acc
template <int MODE>
__global__ __launch_bounds__(256) void mm_bf16_kernel(
    const __bf16* __restrict__ Ag, long long aStride,
    const __bf16* __restrict__ Bg, long long bStride,
    int K, int Nn,
    const float* __restrict__ dinvAll, int dinvStride,
    const float* __restrict__ bias,
    __bf16* __restrict__ Cg, long long cStride,
    float* __restrict__ Cf, long long cfStride) {
  const int b = blockIdx.z;
  const __bf16* A  = Ag + (long long)b * aStride;
  const __bf16* Bm = Bg + (long long)b * bStride;
  const int n0 = blockIdx.x * BN;
  const int m0 = blockIdx.y * BM;
  const int tid  = threadIdx.x;
  const int lane = tid & 31;
  const int wave = tid >> 5;
  const int wm = wave >> 1;   // 0..3 -> 32 rows
  const int wn = wave & 1;    // 0..1 -> 64 cols

  __shared__ __bf16 As[2][BM][BK + LDSPAD];
  __shared__ __bf16 Bs[2][BN][BK + LDSPAD];

  v8f acc[2][4] = {};

  // cooperative load mapping
  const int am   = tid >> 1;            // 0..127  (A row)
  const int aseg = (tid & 1) * 16;      // 0 / 16  (A col segment)
  const int bk   = tid >> 3;            // 0..31   (B row = k)
  const int bn   = (tid & 7) * 16;      // 0..112  (B col segment)

  const int lm  = lane & 15;
  const int kbA = (lane >> 4) * 8;      // A frag: chunks at kbA, kbA+16
  const int kbB = (lane >> 4) * 8;      // B frag: chunks at kbB, kbB+16

  auto load_tiles = [&](int k0, int buf) {
    // A tile: 128 x 32, two async b128 copies per thread
    const __bf16* asrc = A + (long long)(m0 + am) * K + k0 + aseg;
    copy16_to_lds(asrc, &As[buf][am][aseg]);
    copy16_to_lds(asrc + 8, &As[buf][am][aseg + 8]);
    // B tile: 32 x 128, transposed into N-major LDS
    const __bf16* bsrc = Bm + (long long)(k0 + bk) * Nn + n0 + bn;
    uint4 b0 = *(const uint4*)bsrc;
    uint4 b1 = *(const uint4*)(bsrc + 8);
    __bf16 btmp[16];
    *(uint4*)&btmp[0] = b0;
    *(uint4*)&btmp[8] = b1;
#pragma unroll
    for (int j = 0; j < 16; ++j) Bs[buf][bn + j][bk] = btmp[j];
  };

  const int T = K / BK;
  load_tiles(0, 0);

  for (int it = 0; it < T; ++it) {
    const int buf = it & 1;
    wait_async_copies();      // my async A-copies done
    __syncthreads();          // everyone's copies/stores visible
    if (it + 1 < T) load_tiles((it + 1) * BK, buf ^ 1);

    v16bf afrag[2], bfrag[4];
#pragma unroll
    for (int mi = 0; mi < 2; ++mi) {
      int r = wm * 32 + mi * 16 + lm;
      v8bf lo = *(const v8bf*)&As[buf][r][kbA];
      v8bf hi = *(const v8bf*)&As[buf][r][kbA + 16];
      afrag[mi] = __builtin_shufflevector(lo, hi, 0, 1, 2, 3, 4, 5, 6, 7, 8, 9,
                                          10, 11, 12, 13, 14, 15);
    }
#pragma unroll
    for (int ni = 0; ni < 4; ++ni) {
      int r = wn * 64 + ni * 16 + lm;
      v8bf lo = *(const v8bf*)&Bs[buf][r][kbB];
      v8bf hi = *(const v8bf*)&Bs[buf][r][kbB + 16];
      bfrag[ni] = __builtin_shufflevector(lo, hi, 0, 1, 2, 3, 4, 5, 6, 7, 8, 9,
                                          10, 11, 12, 13, 14, 15);
    }
#pragma unroll
    for (int mi = 0; mi < 2; ++mi)
#pragma unroll
      for (int ni = 0; ni < 4; ++ni)
        acc[mi][ni] = __builtin_amdgcn_wmma_f32_16x16x32_bf16(
            false, afrag[mi], false, bfrag[ni], (short)0, acc[mi][ni], false,
            false);
  }

  // epilogue: lane l, vgpr r -> (M = 8*(l>=16)+r, N = l&15)
  const int cn  = lane & 15;
  const int cm0 = (lane >> 4) * 8;
  const float* dinv = (MODE != 0) ? (dinvAll + b * dinvStride) : nullptr;

#pragma unroll
  for (int mi = 0; mi < 2; ++mi) {
#pragma unroll
    for (int ni = 0; ni < 4; ++ni) {
      const int gmb = m0 + wm * 32 + mi * 16 + cm0;
      const int gn  = n0 + wn * 64 + ni * 16 + cn;
#pragma unroll
      for (int r = 0; r < 8; ++r) {
        float v = acc[mi][ni][r];
        if (MODE == 0) {
          Cg[(long long)b * cStride + (long long)(gmb + r) * Nn + gn] = (__bf16)v;
        } else if (MODE == 1) {
          float o = dinv[gmb + r] * v + bias[gn];
          o = o > 0.0f ? o : 0.0f;
          Cg[(long long)b * cStride + (long long)(gmb + r) * Nn + gn] = (__bf16)o;
        } else {
          Cf[(long long)b * cfStride + (long long)(gmb + r) * Nn + gn] =
              dinv[gmb + r] * v;
        }
      }
    }
  }
}

// ---------------- launcher ----------------
extern "C" void kernel_launch(void* const* d_in, const int* in_sizes, int n_in,
                              void* d_out, int out_size, void* d_ws,
                              size_t ws_size, hipStream_t stream) {
  const float* x   = (const float*)d_in[0];
  const int*   adj = (const int*)d_in[1];
  const float* W1  = (const float*)d_in[2];
  const float* b1  = (const float*)d_in[3];
  const float* W2  = (const float*)d_in[4];
  const float* b2  = (const float*)d_in[5];
  float* out = (float*)d_out;

  char* ws = (char*)d_ws;
  size_t off = 0;
  auto carve = [&](size_t bytes) {
    void* p = ws + off;
    off = (off + bytes + 255) & ~(size_t)255;
    return p;
  };
  float*  dinv = (float*)carve((size_t)kBatch * kN * sizeof(float));
  __bf16* ats  = (__bf16*)carve((size_t)kBatch * kN * kN * 2);
  __bf16* xbf  = (__bf16*)carve((size_t)kBatch * kN * kFin * 2);
  __bf16* w1bf = (__bf16*)carve((size_t)kFin * kHid * 2);
  __bf16* w2bf = (__bf16*)carve((size_t)kHid * kOut * 2);
  __bf16* hw   = (__bf16*)carve((size_t)kBatch * kN * kHid * 2);
  __bf16* h    = (__bf16*)carve((size_t)kBatch * kN * kHid * 2);
  __bf16* m2   = (__bf16*)carve((size_t)kBatch * kN * kOut * 2);
  float*  out2 = (float*)carve((size_t)kBatch * kN * kOut * sizeof(float));

  // --- prep ---
  {
    int n = kBatch * kN * kFin;
    f32_to_bf16_kernel<<<(n + 255) / 256, 256, 0, stream>>>(x, xbf, n);
  }
  {
    int n = kFin * kHid;
    f32_to_bf16_kernel<<<(n + 255) / 256, 256, 0, stream>>>(W1, w1bf, n);
  }
  {
    int n = kHid * kOut;
    f32_to_bf16_kernel<<<(n + 255) / 256, 256, 0, stream>>>(W2, w2bf, n);
  }
  deg_kernel<<<(kBatch * kN) / 256, 256, 0, stream>>>(adj, dinv);
  build_ats_kernel<<<dim3(kN / 32, kN / 32, kBatch), dim3(32, 8), 0, stream>>>(
      adj, dinv, ats);

  // --- GEMM1: hw = xbf @ w1bf   (M=1024, K=256, N=512 per batch) ---
  mm_bf16_kernel<0><<<dim3(kHid / BN, kN / BM, kBatch), 256, 0, stream>>>(
      xbf, (long long)kN * kFin, w1bf, 0, kFin, kHid, nullptr, 0, nullptr, hw,
      (long long)kN * kHid, nullptr, 0);

  // --- AGG1: h = relu(dinv * (ats @ hw) + b1)  (K=1024, N=512) ---
  mm_bf16_kernel<1><<<dim3(kHid / BN, kN / BM, kBatch), 256, 0, stream>>>(
      ats, (long long)kN * kN, hw, (long long)kN * kHid, kN, kHid, dinv, kN,
      b1, h, (long long)kN * kHid, nullptr, 0);

  // --- GEMM2: m2 = h @ w2bf   (K=512, N=256) ---
  mm_bf16_kernel<0><<<dim3(kOut / BN, kN / BM, kBatch), 256, 0, stream>>>(
      h, (long long)kN * kHid, w2bf, 0, kHid, kOut, nullptr, 0, nullptr, m2,
      (long long)kN * kOut, nullptr, 0);

  // --- AGG2: out2 = dinv * (ats @ m2)   (K=1024, N=256), f32 ---
  mm_bf16_kernel<2><<<dim3(kOut / BN, kN / BM, kBatch), 256, 0, stream>>>(
      ats, (long long)kN * kN, m2, (long long)kN * kOut, kN, kOut, dinv, kN,
      nullptr, nullptr, 0, out2, (long long)kN * kOut);

  // --- mean pool + bias ---
  pool_kernel<<<(kBatch * kOut) / 256, 256, 0, stream>>>(out2, b2, out);
}
